// RobertaSelfAttention_83906481094903
// MI455X (gfx1250) — matrix operand order repaired and checked by reading, not compiled, not run on hardware
//
#include <hip/hip_runtime.h>
#include <hip/hip_bf16.h>

typedef __attribute__((ext_vector_type(16))) _Float16 v16h;
typedef __attribute__((ext_vector_type(8)))  _Float16 h8;
typedef __attribute__((ext_vector_type(2)))  _Float16 h2;
typedef __attribute__((ext_vector_type(8)))  float    v8f;
typedef __attribute__((ext_vector_type(4)))  unsigned int v4u;
typedef __attribute__((ext_vector_type(8)))  int      v8i;
typedef __attribute__((ext_vector_type(4)))  int      v4i;

#define BATCH 2
#define SEQ   2048
#define DMODEL 1024
#define NHEAD 16
#define HDIM  64

__device__ __forceinline__ v8f wmma_f16(v16h a, v16h b, v8f c) {
    return __builtin_amdgcn_wmma_f32_16x16x32_f16(
        false, a, false, b, (short)0, c, false, false);
}

// 16-bit A-matrix 16x32: lane holds row M=lane&15; pair e=2v:
// k = (v<4?0:16) + half*8 + 2*(v&3)
__device__ __forceinline__ int a_kbase(int v, int half) {
    return ((v & 4) ? 16 : 0) + half * 8 + ((v & 3) << 1);
}
// 16-bit B-matrix 32x16: lane holds column N=lane&15; pair e=2v:
// k = (v<4?0:16) + half*8 + 2*(v&3)
__device__ __forceinline__ int b_kbase(int v, int half) {
    return ((v & 4) ? 16 : 0) + half * 8 + ((v & 3) << 1);
}

// ---------------------------------------------------------------------------
// TDM: async 2D tile load (tile1 rows x tile0 elems of f16, row stride
// stride0 elems) from global into LDS.  D# per CDNA5 ISA ch.8 bit layout.
// clang-23 toolchain: 6-arg builtin.
// ---------------------------------------------------------------------------
__device__ __forceinline__ void tdm_load_2d(const _Float16* g, unsigned lds_byte_off,
                                            unsigned tile0, unsigned tile1,
                                            unsigned td0, unsigned td1,
                                            unsigned stride0) {
    unsigned long long ga = (unsigned long long)(const void*)g;
    v4u g0;
    g0[0] = 1u;                                           // count=1 (valid), user mode
    g0[1] = lds_byte_off;                                 // lds_addr
    g0[2] = (unsigned)(ga & 0xFFFFFFFFu);                 // global_addr[31:0]
    g0[3] = (unsigned)((ga >> 32) & 0x1FFFFFFu)           // global_addr[56:32]
          | (2u << 30);                                   // type = 2 (image)
    v8i g1;
    g1[0] = (int)(1u << 16);                  // workgroup_mask=0, data_size=1 (2B)
    g1[1] = (int)(td0 << 16);                 // tensor_dim0 lo16 (bits 63:48)
    g1[2] = (int)((td0 >> 16) | (td1 << 16)); // td0 hi | tensor_dim1 lo
    g1[3] = (int)((td1 >> 16) | (tile0 << 16)); // td1 hi | tile_dim0
    g1[4] = (int)tile1;                       // tile_dim1, tile_dim2=0
    g1[5] = (int)stride0;                     // tensor_dim0_stride lo32
    g1[6] = 0;                                // stride0 hi | tensor_dim1_stride lo
    g1[7] = 0;
    v4i gz4 = {0, 0, 0, 0};
    v8i gz8 = {0, 0, 0, 0, 0, 0, 0, 0};
    __builtin_amdgcn_tensor_load_to_lds(g0, g1, gz4, gz4, gz8, 0);
}

__device__ __forceinline__ unsigned lds_off(const void* p) {
    return (unsigned)(unsigned long long)p;   // addr[31:0] = LDS byte offset
}

// ---------------------------------------------------------------------------
// One-time conversions: hidden f32 -> f16, W f32 -> f16 transposed [col][k]
// ---------------------------------------------------------------------------
__global__ __launch_bounds__(256)
void cvt_x_kernel(const float* __restrict__ X, _Float16* __restrict__ Xh) {
    int i = (blockIdx.x * 256 + threadIdx.x) * 2;
    float2 f = *(const float2*)(X + i);
    h2 o; o[0] = (_Float16)f.x; o[1] = (_Float16)f.y;
    *(h2*)(Xh + i) = o;
}

__global__ __launch_bounds__(256)
void cvt_wT_kernel(const float* __restrict__ W, _Float16* __restrict__ WT) {
    __shared__ float tile[16][17];
    int tx = threadIdx.x, ty = threadIdx.y;
    int k0 = blockIdx.y * 16, c0 = blockIdx.x * 16;
    tile[ty][tx] = W[(size_t)(k0 + ty) * DMODEL + c0 + tx];
    __syncthreads();
    WT[(size_t)(c0 + ty) * DMODEL + k0 + tx] = (_Float16)tile[tx][ty];
}

// ---------------------------------------------------------------------------
// QKV projection (f16 operands): one wave -> 16 rows x 64 cols (one head).
// TR=false: out[b,h,l,dh]   (Q, K layout)
// TR=true : out[b,h,dh,l]   (V^T layout; packed 8xf16 stores, l-contiguous)
// grid=(B*L/16, H), block=32
// ---------------------------------------------------------------------------
template <bool TR>
__global__ __launch_bounds__(32)
void qkv_proj_kernel(const _Float16* __restrict__ Xh,   // [B*L, D] f16
                     const _Float16* __restrict__ WT,   // [D(col), D(k)] f16
                     const float* __restrict__ bias,    // [D]
                     _Float16* __restrict__ out)
{
    const int lane  = threadIdx.x;
    const int n     = lane & 15;
    const int half  = lane >> 4;
    const int rtile = blockIdx.x;
    const int head  = blockIdx.y;

    const int arow = rtile * 16 + n;
    const h2* xr = reinterpret_cast<const h2*>(Xh + (size_t)arow * DMODEL);

    v8f acc[4] = {v8f{}, v8f{}, v8f{}, v8f{}};

    for (int kk = 0; kk < DMODEL; kk += 32) {
        v16h a;
#pragma unroll
        for (int v = 0; v < 8; ++v) {
            int kb = a_kbase(v, half);
            h2 p = xr[(kk + kb) >> 1];
            a[2 * v] = p[0]; a[2 * v + 1] = p[1];
        }
#pragma unroll
        for (int t = 0; t < 4; ++t) {
            const int col = head * HDIM + t * 16 + n;
            const h2* wr = reinterpret_cast<const h2*>(WT + (size_t)col * DMODEL);
            v16h bf;
#pragma unroll
            for (int v = 0; v < 8; ++v) {
                int kb = b_kbase(v, half);
                h2 p = wr[(kk + kb) >> 1];
                bf[2 * v] = p[0]; bf[2 * v + 1] = p[1];
            }
            acc[t] = wmma_f16(a, bf, acc[t]);
        }
    }

    const int b  = (rtile * 16) / SEQ;          // whole tile is in one batch
    const int l0 = (rtile * 16) % SEQ;
#pragma unroll
    for (int t = 0; t < 4; ++t) {
        const float bv = bias[head * HDIM + t * 16 + n];
        if (TR) {
            // V^T: lane's 8 acc rows are l-consecutive -> one 16B store
            h8 pack;
#pragma unroll
            for (int r = 0; r < 8; ++r) pack[r] = (_Float16)(acc[t][r] + bv);
            int dh = t * 16 + n;
            size_t idx = ((size_t)(b * NHEAD + head) * HDIM + dh) * SEQ + l0 + 8 * half;
            *(h8*)(out + idx) = pack;
        } else {
#pragma unroll
            for (int r = 0; r < 8; ++r) {
                int l = l0 + r + 8 * half;      // C-layout row mapping
                size_t idx = (((size_t)(b * NHEAD + head) * SEQ + l) * HDIM) + t * 16 + n;
                out[idx] = (_Float16)(acc[t][r] + bv);
            }
        }
    }
}

// ---------------------------------------------------------------------------
// Flash attention: TDM double-buffered K (32x64) and V^T (64x32) staging,
// NT bias streaming.  grid=(L/16, B*H), block=32
// ---------------------------------------------------------------------------
__global__ __launch_bounds__(32)
void attn_kernel(const _Float16* __restrict__ Q,   // [B,H,L,HDIM]
                 const _Float16* __restrict__ K,   // [B,H,L,HDIM]
                 const _Float16* __restrict__ Vt,  // [B,H,HDIM,L]
                 const float* __restrict__ rel1,   // [1,H,L,L]
                 const float* __restrict__ rel2,   // [1,H,L,L]
                 const float* __restrict__ mask,   // [B,1,1,L]
                 float* __restrict__ out)          // [B,L,H*HDIM]
{
    __shared__ __align__(16) _Float16 sK[2][32 * HDIM];   // [m=32][dh=64]
    __shared__ __align__(16) _Float16 sV[2][HDIM * 32];   // [dh=64][m=32]
    __shared__ __align__(16) _Float16 sP[16 * 32];

    const int lane = threadIdx.x;
    const int n    = lane & 15;
    const int half = lane >> 4;
    const int ltile = blockIdx.x;
    const int bh    = blockIdx.y;
    const int b     = bh >> 4;
    const int h     = bh & 15;

    // Q A-fragments (contraction over dh = 64, two steps of 32)
    const int ql = ltile * 16 + n;
    const h2* Qrow = reinterpret_cast<const h2*>(Q + ((size_t)bh * SEQ + ql) * HDIM);
    v16h qa0, qa1;
#pragma unroll
    for (int v = 0; v < 8; ++v) {
        int kb = a_kbase(v, half);
        h2 p0 = Qrow[kb >> 1];
        h2 p1 = Qrow[(32 + kb) >> 1];
        qa0[2 * v] = p0[0]; qa0[2 * v + 1] = p0[1];
        qa1[2 * v] = p1[0]; qa1[2 * v + 1] = p1[1];
    }

    float maxr[8], sumr[8];
#pragma unroll
    for (int r = 0; r < 8; ++r) { maxr[r] = -3.0e38f; sumr[r] = 0.0f; }
    v8f oacc[4] = {v8f{}, v8f{}, v8f{}, v8f{}};

    const _Float16* Kbh = K  + (size_t)bh * SEQ * HDIM;
    const _Float16* Vbh = Vt + (size_t)bh * HDIM * SEQ;
    const unsigned offK[2] = { lds_off(&sK[0][0]), lds_off(&sK[1][0]) };
    const unsigned offV[2] = { lds_off(&sV[0][0]), lds_off(&sV[1][0]) };

    // Prefetch chunk 0: K rows [0,32) x 64; V^T rows 64 x cols [0,32)
    tdm_load_2d(Kbh, offK[0], HDIM, 32, HDIM, SEQ, HDIM);
    tdm_load_2d(Vbh, offV[0], 32, HDIM, SEQ, HDIM, SEQ);

    int ci = 0;
    for (int mc = 0; mc < SEQ; mc += 32, ci ^= 1) {
        if (mc + 32 < SEQ) {
            tdm_load_2d(Kbh + (size_t)(mc + 32) * HDIM, offK[ci ^ 1], HDIM, 32, HDIM, SEQ, HDIM);
            tdm_load_2d(Vbh + (mc + 32),               offV[ci ^ 1], 32, HDIM, SEQ, HDIM, SEQ);
            __builtin_amdgcn_s_wait_tensorcnt(2);   // current chunk's 2 DMAs done
        } else {
            __builtin_amdgcn_s_wait_tensorcnt(0);
        }
        asm volatile("" ::: "memory");
        const _Float16* cK = &sK[ci][0];
        const _Float16* cV = &sV[ci][0];

        // ---- Scores S0 (cols mc..mc+15) and S1 (mc+16..mc+31) ----
        v8f s0 = v8f{}, s1 = v8f{};
#pragma unroll
        for (int step = 0; step < 2; ++step) {
            const h2* kr0 = reinterpret_cast<const h2*>(cK + n * HDIM);
            const h2* kr1 = reinterpret_cast<const h2*>(cK + (16 + n) * HDIM);
            v16h kf0, kf1;
#pragma unroll
            for (int v = 0; v < 8; ++v) {
                int kb = b_kbase(v, half) + step * 32;
                h2 p0 = kr0[kb >> 1];
                h2 p1 = kr1[kb >> 1];
                kf0[2 * v] = p0[0]; kf0[2 * v + 1] = p0[1];
                kf1[2 * v] = p1[0]; kf1[2 * v + 1] = p1[1];
            }
            s0 = wmma_f16(step ? qa1 : qa0, kf0, s0);
            s1 = wmma_f16(step ? qa1 : qa0, kf1, s1);
        }

        // ---- Bias (streamed NT) + scale + mask, online softmax ----
        float sc[8], p0[8], p1[8];
#pragma unroll
        for (int r = 0; r < 8; ++r) {
            int rowg = ltile * 16 + r + 8 * half;
            size_t bb = ((size_t)h * SEQ + rowg) * SEQ;
            int c0 = mc + n, c1 = mc + 16 + n;
            float b10 = __builtin_nontemporal_load(rel1 + bb + c0);
            float b11 = __builtin_nontemporal_load(rel1 + bb + c1);
            float b20 = __builtin_nontemporal_load(rel2 + bb + c0);
            float b21 = __builtin_nontemporal_load(rel2 + bb + c1);
            float t0 = (s0[r] + b10 + b20) * 0.125f + mask[b * SEQ + c0];
            float t1 = (s1[r] + b11 + b21) * 0.125f + mask[b * SEQ + c1];
            float cm = fmaxf(t0, t1);
            cm = fmaxf(cm, __shfl_xor(cm, 1));
            cm = fmaxf(cm, __shfl_xor(cm, 2));
            cm = fmaxf(cm, __shfl_xor(cm, 4));
            cm = fmaxf(cm, __shfl_xor(cm, 8));
            float nm = fmaxf(maxr[r], cm);
            sc[r] = __expf(maxr[r] - nm);
            maxr[r] = nm;
            p0[r] = __expf(t0 - nm);
            p1[r] = __expf(t1 - nm);
            float rs = p0[r] + p1[r];
            rs += __shfl_xor(rs, 1);
            rs += __shfl_xor(rs, 2);
            rs += __shfl_xor(rs, 4);
            rs += __shfl_xor(rs, 8);
            sumr[r] = sumr[r] * sc[r] + rs;
        }
#pragma unroll
        for (int t = 0; t < 4; ++t)
#pragma unroll
            for (int r = 0; r < 8; ++r) oacc[t][r] *= sc[r];

        // ---- P: C-layout -> LDS -> A-fragment ----
#pragma unroll
        for (int r = 0; r < 8; ++r) {
            int prow = r + 8 * half;
            sP[prow * 32 + n]      = (_Float16)p0[r];
            sP[prow * 32 + 16 + n] = (_Float16)p1[r];
        }
        __syncthreads();

        const h2* pr = reinterpret_cast<const h2*>(sP + n * 32);
        v16h pa;
#pragma unroll
        for (int v = 0; v < 8; ++v) {
            int kb = a_kbase(v, half);
            h2 p = pr[kb >> 1];
            pa[2 * v] = p[0]; pa[2 * v + 1] = p[1];
        }

        // ---- O += P @ V : V^T tile is [dh][m], fragments now contiguous ----
#pragma unroll
        for (int t = 0; t < 4; ++t) {
            const h2* vr = reinterpret_cast<const h2*>(cV + (t * 16 + n) * 32);
            v16h vf;
#pragma unroll
            for (int v = 0; v < 8; ++v) {
                int kb = b_kbase(v, half);          // m index within chunk
                h2 p = vr[kb >> 1];
                vf[2 * v] = p[0]; vf[2 * v + 1] = p[1];
            }
            oacc[t] = wmma_f16(pa, vf, oacc[t]);
        }
        __syncthreads();   // sP reuse next iteration
    }

    // ---- Normalize + NT store ctx[b, l, h*64+dh] ----
#pragma unroll
    for (int t = 0; t < 4; ++t) {
#pragma unroll
        for (int r = 0; r < 8; ++r) {
            int rowg = ltile * 16 + r + 8 * half;
            size_t idx = ((size_t)b * SEQ + rowg) * DMODEL + h * HDIM + t * 16 + n;
            __builtin_nontemporal_store(oacc[t][r] / sumr[r], out + idx);
        }
    }
}

extern "C" void kernel_launch(void* const* d_in, const int* in_sizes, int n_in,
                              void* d_out, int out_size, void* d_ws, size_t ws_size,
                              hipStream_t stream) {
    const float* hidden = (const float*)d_in[0];
    const float* amask  = (const float*)d_in[1];
    const float* rel1   = (const float*)d_in[2];
    const float* rel2   = (const float*)d_in[3];
    const float* Wq     = (const float*)d_in[4];
    const float* bq     = (const float*)d_in[5];
    const float* Wk     = (const float*)d_in[6];
    const float* bk     = (const float*)d_in[7];
    const float* Wv     = (const float*)d_in[8];
    const float* bv     = (const float*)d_in[9];
    float* out = (float*)d_out;

    const size_t QN = (size_t)BATCH * NHEAD * SEQ * HDIM;   // 4,194,304
    const size_t XN = (size_t)BATCH * SEQ * DMODEL;         // 4,194,304
    const size_t WN = (size_t)DMODEL * DMODEL;              // 1,048,576
    _Float16* Qh  = (_Float16*)d_ws;
    _Float16* Kh  = Qh + QN;
    _Float16* Vth = Kh + QN;
    _Float16* Xh  = Vth + QN;
    _Float16* WqT = Xh + XN;
    _Float16* WkT = WqT + WN;
    _Float16* WvT = WkT + WN;

    cvt_x_kernel<<<dim3(XN / 512), dim3(256), 0, stream>>>(hidden, Xh);
    dim3 tgrid(DMODEL / 16, DMODEL / 16), tblk(16, 16);
    cvt_wT_kernel<<<tgrid, tblk, 0, stream>>>(Wq, WqT);
    cvt_wT_kernel<<<tgrid, tblk, 0, stream>>>(Wk, WkT);
    cvt_wT_kernel<<<tgrid, tblk, 0, stream>>>(Wv, WvT);

    dim3 pgrid(BATCH * SEQ / 16, NHEAD), pblk(32);
    qkv_proj_kernel<false><<<pgrid, pblk, 0, stream>>>(Xh, WqT, bq, Qh);
    qkv_proj_kernel<false><<<pgrid, pblk, 0, stream>>>(Xh, WkT, bk, Kh);
    qkv_proj_kernel<true ><<<pgrid, pblk, 0, stream>>>(Xh, WvT, bv, Vth);

    dim3 agrid(SEQ / 16, BATCH * NHEAD);
    attn_kernel<<<agrid, pblk, 0, stream>>>(Qh, Kh, Vth, rel1, rel2, amask, out);
}
